// FastRerank_55980603736133
// MI455X (gfx1250) — matrix operand order repaired and checked by reading, not compiled, not run on hardware
//
#include <hip/hip_runtime.h>
#include <hip/hip_bf16.h>
#include <math.h>

typedef __attribute__((ext_vector_type(16))) __bf16 v16bf;
typedef __attribute__((ext_vector_type(8)))  float  v8f;

#define NBLOCKS 11
#define Dm 128
#define Km 7
#define Bb 8
#define LAa 1024
#define LTt 64
#define TOPKk 10
#define KTOT (Dm*Km)      /* 896 */
#define KSTEPS (KTOT/32)  /* 28  */
#define OC 256
#define LDSB_STRIDE 912   /* 896 + 16 pad: breaks LDS bank alignment */

// ---------------------------------------------------------------------------
// Prepack conv weights into bf16 WMMA A-fragment lane order.
// Layout: wp[((blk*16 + ocTile)*28 + kStep)*512 + lane*16 + e]
// A 16x32 bf16 layout (contiguous k): lane<16: e0..15->K0..15;
// lane>=16: e0..15->K16..31.  K index kk = tap*128 + ic.
// ---------------------------------------------------------------------------
__global__ __launch_bounds__(32) void prepack_w(const float* __restrict__ conv_w,
                                                __bf16* __restrict__ wp) {
    int wg   = blockIdx.x;            // ((blk*16 + tile)*28 + step)
    int step = wg % KSTEPS;
    int tmp  = wg / KSTEPS;
    int tile = tmp % 16;
    int blk  = tmp / 16;
    int lane = threadIdx.x;
    int m = lane & 15, half = lane >> 4;
    int oc = tile * 16 + m;
    __bf16* dst = wp + (size_t)wg * 512 + lane * 16;
#pragma unroll
    for (int e = 0; e < 16; ++e) {
        int kl  = e + 16 * half;
        int kk  = step * 32 + kl;
        int tap = kk >> 7;            // kk / 128
        int ic  = kk & 127;
        float v = conv_w[(((size_t)blk * OC + oc) * Dm + ic) * Km + tap];
        dst[e] = (__bf16)v;
    }
}

// ---------------------------------------------------------------------------
// Embedding gather: h[pos][d] = emb[words[pos]][d]   (fp32 residual stream)
// ---------------------------------------------------------------------------
__global__ void embed_gather(const int* __restrict__ words,
                             const float* __restrict__ emb,
                             float* __restrict__ h, int total /* B*L */) {
    int idx = blockIdx.x * blockDim.x + threadIdx.x;
    if (idx >= total * Dm) return;
    int pos = idx >> 7, d = idx & 127;
    int w = words[pos];
    h[idx] = emb[(size_t)w * Dm + d];
}

// ---------------------------------------------------------------------------
// One dilated conv block + GLU + residual, fused.
// Grid: (L/16, B), block 256 (8 wave32s). Each WG: 16 positions, all 256 oc.
// im2col tile staged in LDS as bf16; 28 WMMA K-steps; y staged in LDS f32;
// GLU+residual written to hOut (double buffered vs hIn to avoid halo races).
// ---------------------------------------------------------------------------
__global__ __launch_bounds__(256) void conv_glu(
        const float* __restrict__ hIn, float* __restrict__ hOut,
        const __bf16* __restrict__ wp,      // this block's packed weights
        const float* __restrict__ bias,     // 256
        int L, int dil) {
    __shared__ __bf16 ldsB[16 * LDSB_STRIDE];   // im2col: [pos][kk], ~28.5 KB
    __shared__ float  ldsY[OC * 16];            // y:      [oc][pos], 16 KB

    int tid = threadIdx.x;
    int tileBase = blockIdx.x * 16;
    int b = blockIdx.y;

    // Stage im2col tile (896*16 = 14336 elements, 56 per thread, coalesced over ic)
    for (int it = 0; it < 56; ++it) {
        int idx = tid + it * 256;
        int p   = idx / KTOT;
        int kk  = idx - p * KTOT;
        int tap = kk >> 7, ic = kk & 127;
        int pos = tileBase + p + (tap - 3) * dil;
        float v = 0.f;
        if (pos >= 0 && pos < L)
            v = hIn[((size_t)(b * L + pos)) * Dm + ic];
        ldsB[p * LDSB_STRIDE + kk] = (__bf16)v;
    }
    __syncthreads();

    int lane = tid & 31, wave = tid >> 5;
    int half = lane >> 4, nn = lane & 15;

    v8f acc0 = {}, acc1 = {};
    const __bf16* wp0 = wp + ((size_t)(2 * wave)     * KSTEPS) * 512 + lane * 16;
    const __bf16* wp1 = wp + ((size_t)(2 * wave + 1) * KSTEPS) * 512 + lane * 16;
    const __bf16* bb  = &ldsB[nn * LDSB_STRIDE + 16 * half];

    for (int s = 0; s < KSTEPS; ++s) {
        v16bf a0 = *(const v16bf*)(wp0 + (size_t)s * 512);
        v16bf a1 = *(const v16bf*)(wp1 + (size_t)s * 512);
        v16bf bf = *(const v16bf*)(bb + s * 32);   // B 32x16: lane<16 K0..15, lane>=16 K16..31
        acc0 = __builtin_amdgcn_wmma_f32_16x16x32_bf16(false, a0, false, bf,
                                                       (short)0, acc0, false, false);
        acc1 = __builtin_amdgcn_wmma_f32_16x16x32_bf16(false, a1, false, bf,
                                                       (short)0, acc1, false, false);
    }

    // C/D layout: lane l, vgpr r -> row = r + 8*(l>>4), col = l&15
#pragma unroll
    for (int r = 0; r < 8; ++r) {
        int row = r + 8 * half;
        ldsY[((2 * wave)     * 16 + row) * 16 + nn] = acc0[r];
        ldsY[((2 * wave + 1) * 16 + row) * 16 + nn] = acc1[r];
    }
    __syncthreads();

    // GLU + residual: h' = h + a * sigmoid(g); a = y[0:128], g = y[128:256]
#pragma unroll
    for (int it = 0; it < 8; ++it) {
        int idx = tid + it * 256;
        int p  = idx >> 7;     // position in tile
        int ic = idx & 127;    // channel (contiguous across threads -> coalesced)
        float a = ldsY[ic * 16 + p] + bias[ic];
        float g = ldsY[(ic + 128) * 16 + p] + bias[128 + ic];
        size_t gi = ((size_t)(b * L + tileBase + p)) * Dm + ic;
        hOut[gi] = hIn[gi] + a * (1.f / (1.f + __expf(-g)));
    }
}

// ---------------------------------------------------------------------------
// Similarity + row max: s[i,j] = exp(-(||a||^2+||t||^2-2 a.t)) * mA * mT
// Grid (LA/16, B), one wave per 16 article rows; 4 N-tiles x 4 K-steps WMMA.
// ---------------------------------------------------------------------------
__global__ __launch_bounds__(32) void sim_rowmax(
        const float* __restrict__ hA, const float* __restrict__ hT,
        const float* __restrict__ mA, const float* __restrict__ mT,
        float* __restrict__ rowmax) {
    __shared__ float na[16];
    __shared__ float ntl[64];
    __shared__ float sT[16 * 64];
    int l = threadIdx.x;
    int b = blockIdx.y;
    int rowBase = blockIdx.x * 16;

    if (l < 16) {
        const float* p = hA + ((size_t)(b * LAa + rowBase + l)) * Dm;
        float s = 0.f;
        for (int d = 0; d < Dm; ++d) s += p[d] * p[d];
        na[l] = s;
    }
    for (int j = l; j < 64; j += 32) {
        const float* p = hT + ((size_t)(b * LTt + j)) * Dm;
        float s = 0.f;
        for (int d = 0; d < Dm; ++d) s += p[d] * p[d];
        ntl[j] = s;
    }
    __syncthreads();

    int m = l & 15, half = l >> 4;
    v16bf afr[4];
#pragma unroll
    for (int s = 0; s < 4; ++s) {
#pragma unroll
        for (int e = 0; e < 16; ++e) {
            int kl = e + 16 * half;
            int ic = s * 32 + kl;
            afr[s][e] = (__bf16)hA[((size_t)(b * LAa + rowBase + m)) * Dm + ic];
        }
    }
    for (int ntile = 0; ntile < 4; ++ntile) {
        int n = ntile * 16 + m;
        v8f acc = {};
#pragma unroll
        for (int s = 0; s < 4; ++s) {
            v16bf bfr;
#pragma unroll
            for (int e = 0; e < 16; ++e) {
                int ic = s * 32 + 16 * half + e;
                bfr[e] = (__bf16)hT[((size_t)(b * LTt + n)) * Dm + ic];
            }
            acc = __builtin_amdgcn_wmma_f32_16x16x32_bf16(false, afr[s], false, bfr,
                                                          (short)0, acc, false, false);
        }
#pragma unroll
        for (int r = 0; r < 8; ++r) {
            int mm = r + 8 * half;
            float d2 = na[mm] + ntl[n] - 2.f * acc[r];
            float sv = __expf(-d2) * mA[b * LAa + rowBase + mm] * mT[b * LTt + n];
            sT[mm * 64 + n] = sv;
        }
    }
    __syncthreads();
    if (l < 16) {
        float mx = sT[l * 64];
        for (int j = 1; j < 64; ++j) mx = fmaxf(mx, sT[l * 64 + j]);
        rowmax[(size_t)b * LAa + rowBase + l] = mx;
    }
}

// ---------------------------------------------------------------------------
// Top-10 (descending, lowest-index ties, matching lax.top_k) + 10x10 MLP.
// ---------------------------------------------------------------------------
__global__ __launch_bounds__(256) void topk_mlp(
        const float* __restrict__ rowmax,
        const float* __restrict__ w1, const float* __restrict__ b1,
        const float* __restrict__ w2, const float* __restrict__ b2,
        float* __restrict__ out) {
    __shared__ float vals[LAa];
    __shared__ float rv[256];
    __shared__ int   ri[256];
    __shared__ float top[TOPKk];
    int t = threadIdx.x;
    int b = blockIdx.x;
    for (int i = t; i < LAa; i += 256) vals[i] = rowmax[(size_t)b * LAa + i];
    __syncthreads();
    for (int k = 0; k < TOPKk; ++k) {
        float best = -INFINITY; int bi = LAa;
        for (int i = t; i < LAa; i += 256) {
            float v = vals[i];
            if (v > best) { best = v; bi = i; }
        }
        rv[t] = best; ri[t] = bi;
        __syncthreads();
        for (int s = 128; s > 0; s >>= 1) {
            if (t < s) {
                float ov = rv[t + s]; int oi = ri[t + s];
                if (ov > rv[t] || (ov == rv[t] && oi < ri[t])) { rv[t] = ov; ri[t] = oi; }
            }
            __syncthreads();
        }
        if (t == 0) { top[k] = rv[0]; vals[ri[0]] = -INFINITY; }
        __syncthreads();
    }
    if (t == 0) {
        float o = b2[0];
        for (int j = 0; j < TOPKk; ++j) {
            float h = b1[j];
            for (int kk = 0; kk < TOPKk; ++kk) h += top[kk] * w1[j * TOPKk + kk];
            h = fmaxf(h, 0.f);
            o += h * w2[j];
        }
        out[b] = o;
    }
}

// ---------------------------------------------------------------------------
extern "C" void kernel_launch(void* const* d_in, const int* in_sizes, int n_in,
                              void* d_out, int out_size, void* d_ws, size_t ws_size,
                              hipStream_t stream) {
    const int*   article_word = (const int*)d_in[0];
    const int*   title_word   = (const int*)d_in[1];
    const float* article_mask = (const float*)d_in[2];
    const float* title_mask   = (const float*)d_in[3];
    const float* emb          = (const float*)d_in[4];
    const float* conv_w       = (const float*)d_in[5];
    const float* conv_b       = (const float*)d_in[6];
    const float* w1           = (const float*)d_in[7];
    const float* b1           = (const float*)d_in[8];
    const float* w2           = (const float*)d_in[9];
    const float* b2           = (const float*)d_in[10];
    float* out = (float*)d_out;

    char* ws = (char*)d_ws;
    size_t off = 0;
    auto alloc = [&](size_t bytes) -> char* {
        char* p = ws + off;
        off += (bytes + 255) & ~(size_t)255;
        return p;
    };
    float*  hA0  = (float*)alloc(sizeof(float) * (size_t)Bb * LAa * Dm);
    float*  hA1  = (float*)alloc(sizeof(float) * (size_t)Bb * LAa * Dm);
    float*  hT0  = (float*)alloc(sizeof(float) * (size_t)Bb * LTt * Dm);
    float*  hT1  = (float*)alloc(sizeof(float) * (size_t)Bb * LTt * Dm);
    __bf16* wp   = (__bf16*)alloc(sizeof(__bf16) * (size_t)NBLOCKS * 16 * KSTEPS * 512);
    float*  rmax = (float*)alloc(sizeof(float) * (size_t)Bb * LAa);

    // 1) prepack weights to bf16 WMMA fragment order (shared by both encoders)
    prepack_w<<<NBLOCKS * 16 * KSTEPS, 32, 0, stream>>>(conv_w, wp);

    // 2) embedding gathers
    embed_gather<<<(Bb * LAa * Dm + 255) / 256, 256, 0, stream>>>(article_word, emb, hA0, Bb * LAa);
    embed_gather<<<(Bb * LTt * Dm + 255) / 256, 256, 0, stream>>>(title_word, emb, hT0, Bb * LTt);

    // 3) 11 conv/GLU/residual blocks, double buffered
    static const int dils[NBLOCKS] = {1, 2, 4, 8, 16, 32, 1, 1, 1, 1, 1};
    float* aIn = hA0; float* aOut = hA1;
    float* tIn = hT0; float* tOut = hT1;
    for (int blk = 0; blk < NBLOCKS; ++blk) {
        const __bf16* wpb = wp + (size_t)blk * 16 * KSTEPS * 512;
        const float*  bb  = conv_b + blk * OC;
        conv_glu<<<dim3(LAa / 16, Bb), 256, 0, stream>>>(aIn, aOut, wpb, bb, LAa, dils[blk]);
        conv_glu<<<dim3(LTt / 16, Bb), 256, 0, stream>>>(tIn, tOut, wpb, bb, LTt, dils[blk]);
        float* tmp;
        tmp = aIn; aIn = aOut; aOut = tmp;
        tmp = tIn; tIn = tOut; tOut = tmp;
    }

    // 4) similarity + row max (WMMA dot products), 5) top-10 + MLP
    sim_rowmax<<<dim3(LAa / 16, Bb), 32, 0, stream>>>(aIn, tIn, article_mask, title_mask, rmax);
    topk_mlp<<<Bb, 256, 0, stream>>>(rmax, w1, b1, w2, b2, out);
}
